// SWCATransformerBlock_62886911148183
// MI455X (gfx1250) — compile-verified, matrix operand order checked by reading, not run on hardware
//
#include <hip/hip_runtime.h>
#include <hip/hip_bf16.h>
#include <math.h>

typedef __attribute__((ext_vector_type(16))) _Float16 v16h;
typedef __attribute__((ext_vector_type(8)))  _Float16 v8h;
typedef __attribute__((ext_vector_type(8)))  float    v8f;
typedef int v4i __attribute__((vector_size(16)));   // matches builtin param type

#define DIMC   256
#define HH     128
#define WWID   128
#define LTOK   (HH*WWID)
#define NHEADS 8
#define DHEAD  32
#define WIN    8
#define NTOK   64
#define HIDD   1024
#define BATCH  4
#define MROWS  (BATCH*LTOK)   // 65536 tokens

// ---------------------------------------------------------------------------
// CDNA5 async global->LDS copy (ASYNCcnt path), with sync fallback
// ---------------------------------------------------------------------------
#if defined(__has_builtin)
#if __has_builtin(__builtin_amdgcn_global_load_async_to_lds_b128) && \
    __has_builtin(__builtin_amdgcn_s_wait_asynccnt)
#define HAVE_ASYNC_LDS 1
#endif
#endif
#ifndef HAVE_ASYNC_LDS
#define HAVE_ASYNC_LDS 0
#endif

__device__ __forceinline__ void async_cp16(const _Float16* gsrc, _Float16* ldst) {
#if HAVE_ASYNC_LDS
    __builtin_amdgcn_global_load_async_to_lds_b128(
        (__attribute__((address_space(1))) v4i*)gsrc,
        (__attribute__((address_space(3))) v4i*)ldst, 0, 0);
#else
    *(v8h*)ldst = *(const v8h*)gsrc;
#endif
}
__device__ __forceinline__ void async_wait0() {
#if HAVE_ASYNC_LDS
    __builtin_amdgcn_s_wait_asynccnt(0);
#endif
}

// ---------------------------------------------------------------------------
// fp32 -> fp16 weight conversion
// ---------------------------------------------------------------------------
__global__ void k_f32_to_f16(const float* __restrict__ in, _Float16* __restrict__ out, int n) {
    int i = blockIdx.x * 256 + threadIdx.x;
    if (i < n) out[i] = (_Float16)in[i];
}

// ---------------------------------------------------------------------------
// LayerNorm over C=256, one wave32 per token, output f16
// ---------------------------------------------------------------------------
__global__ __launch_bounds__(256) void k_ln(const float* __restrict__ x,
                                            const float* __restrict__ g,
                                            const float* __restrict__ b,
                                            _Float16* __restrict__ out) {
    int wave = threadIdx.x >> 5, lane = threadIdx.x & 31;
    size_t row = (size_t)blockIdx.x * 8 + wave;
    const float* p = x + row * DIMC;
    float v[8];
    float s = 0.f;
#pragma unroll
    for (int i = 0; i < 8; i++) { v[i] = p[lane + 32 * i]; s += v[i]; }
#pragma unroll
    for (int o = 16; o > 0; o >>= 1) s += __shfl_xor(s, o, 32);
    float mean = s * (1.f / 256.f);
    float vs = 0.f;
#pragma unroll
    for (int i = 0; i < 8; i++) { float d = v[i] - mean; vs += d * d; }
#pragma unroll
    for (int o = 16; o > 0; o >>= 1) vs += __shfl_xor(vs, o, 32);
    float rstd = rsqrtf(vs * (1.f / 256.f) + 1e-5f);
    _Float16* q = out + row * DIMC;
#pragma unroll
    for (int i = 0; i < 8; i++) {
        int c = lane + 32 * i;
        q[c] = (_Float16)((v[i] - mean) * rstd * g[c] + b[c]);
    }
}

// ---------------------------------------------------------------------------
// 16x32 f16 WMMA operand from row-major LDS:
// lane holds row = lane%16; K chunks [hl*8 .. +7] and [16+hl*8 .. +7]
// ---------------------------------------------------------------------------
__device__ __forceinline__ v16h ldfrag(const _Float16* p, int hl) {
    v8h lo = *(const v8h*)(p + hl * 8);
    v8h hi = *(const v8h*)(p + 16 + hl * 8);
    v16h f;
#pragma unroll
    for (int e = 0; e < 8; e++) { f[e] = lo[e]; f[e + 8] = hi[e]; }
    return f;
}

// ---------------------------------------------------------------------------
// WMMA GEMM: out[M,N] = A[M,K](f16) * W[N,K](f16)^T + bias (+resid)
// 128x128 block tile, 8 waves (2x4), each wave 64x32 = 4x2 wmma tiles.
// Ping-pong LDS; next tile staged via async global->LDS right after the
// barrier (its previous readers are already done); wait_asynccnt + barrier
// at iteration end. One barrier per K-step.
// ---------------------------------------------------------------------------
#define TBM 128
#define TBN 128
#define TBK 32
#define LDST 40          // padded f16 row stride (80B, 16B aligned)

template <bool OUTF16, bool RESID>
__global__ __launch_bounds__(256) void k_gemm(const _Float16* __restrict__ A,
                                              const _Float16* __restrict__ W,
                                              const float* __restrict__ bias,
                                              void* __restrict__ outp,
                                              const float* __restrict__ resid,
                                              int M, int N, int K) {
    // [2 bufs][A|W][128 rows][40] f16  = 40 KB total
    __shared__ _Float16 smem[2 * 2 * TBM * LDST];
#define SA(p, r, c) smem[((((p)*2 + 0) * TBM) + (r)) * LDST + (c)]
#define SW(p, r, c) smem[((((p)*2 + 1) * TBM) + (r)) * LDST + (c)]

    int t = threadIdx.x, lane = t & 31, wave = t >> 5;
    int wm = (wave >> 2) * 64;       // 0 or 64
    int wn = (wave & 3) * 32;        // 0,32,64,96
    int mrow0 = blockIdx.y * TBM, nrow0 = blockIdx.x * TBN;
    int hl = lane >> 4, mr = lane & 15;

    int lr = t >> 1;                 // staging row 0..127
    int lc = (t & 1) * 16;           // staging col 0 or 16
    const _Float16* gA = A + (size_t)(mrow0 + lr) * K + lc;
    const _Float16* gW = W + (size_t)(nrow0 + lr) * K + lc;

    v8f acc[4][2] = {};

    auto stage = [&](int buf, int k) {
        async_cp16(gA + k,     &SA(buf, lr, lc));
        async_cp16(gA + k + 8, &SA(buf, lr, lc + 8));
        async_cp16(gW + k,     &SW(buf, lr, lc));
        async_cp16(gW + k + 8, &SW(buf, lr, lc + 8));
    };

    stage(0, 0);
    async_wait0();
    __syncthreads();

    int p = 0;
    for (int k0 = 0; k0 < K; k0 += TBK) {
        bool has_next = (k0 + TBK) < K;
        if (has_next) stage(p ^ 1, k0 + TBK);
        if (k0 + 2 * TBK < K) {
            __builtin_prefetch(gA + k0 + 2 * TBK, 0, 1);
            __builtin_prefetch(gW + k0 + 2 * TBK, 0, 1);
        }
        v16h af[4], bf[2];
#pragma unroll
        for (int i = 0; i < 4; i++) af[i] = ldfrag(&SA(p, wm + i * 16 + mr, 0), hl);
#pragma unroll
        for (int j = 0; j < 2; j++) bf[j] = ldfrag(&SW(p, wn + j * 16 + mr, 0), hl);
#pragma unroll
        for (int i = 0; i < 4; i++)
#pragma unroll
            for (int j = 0; j < 2; j++)
                acc[i][j] = __builtin_amdgcn_wmma_f32_16x16x32_f16(
                    false, af[i], false, bf[j], (short)0, acc[i][j], false, false);
        if (has_next) {
            async_wait0();
            __syncthreads();
            p ^= 1;
        }
    }

    if (OUTF16) {
        // LDS-transpose epilogue: wave-private 64x40 region, then b128 stores
        __syncthreads();
        _Float16* ep = smem + wave * 64 * LDST;
#pragma unroll
        for (int i = 0; i < 4; i++)
#pragma unroll
            for (int j = 0; j < 2; j++) {
                float bv = bias[nrow0 + wn + j * 16 + mr];
#pragma unroll
                for (int r = 0; r < 8; r++)
                    ep[(i * 16 + hl * 8 + r) * LDST + j * 16 + mr] =
                        (_Float16)(acc[i][j][r] + bv);
            }
        _Float16* outH = (_Float16*)outp;
#pragma unroll
        for (int rr = 0; rr < 2; rr++) {
            int lrow = lane * 2 + rr;
            size_t gm = (size_t)(mrow0 + wm + lrow);
            const _Float16* src = ep + lrow * LDST;
            v8h c0 = *(const v8h*)(src);
            v8h c1 = *(const v8h*)(src + 8);
            v8h c2 = *(const v8h*)(src + 16);
            v8h c3 = *(const v8h*)(src + 24);
            _Float16* dst = outH + gm * N + nrow0 + wn;
            *(v8h*)(dst)      = c0;
            *(v8h*)(dst + 8)  = c1;
            *(v8h*)(dst + 16) = c2;
            *(v8h*)(dst + 24) = c3;
        }
    } else {
        float* outF = (float*)outp;
#pragma unroll
        for (int i = 0; i < 4; i++)
#pragma unroll
            for (int j = 0; j < 2; j++) {
                int gn = nrow0 + wn + j * 16 + mr;
                float bv = bias[gn];
#pragma unroll
                for (int r = 0; r < 8; r++) {
                    int gm = mrow0 + wm + i * 16 + hl * 8 + r;
                    size_t idx = (size_t)gm * N + gn;
                    float v = acc[i][j][r] + bv;
                    if (RESID) v += resid[idx];
                    outF[idx] = v;
                }
            }
    }
#undef SA
#undef SW
}

// ---------------------------------------------------------------------------
// Fused window attention: one wave32 per (window, head).
// ---------------------------------------------------------------------------
__global__ __launch_bounds__(32) void k_attn(const _Float16* __restrict__ qkv,
                                             const float* __restrict__ dww,
                                             const float* __restrict__ dwb,
                                             const float* __restrict__ rpb,
                                             const float* __restrict__ temp,
                                             _Float16* __restrict__ attno) {
    __shared__ _Float16 raw[3][NTOK][DHEAD];   // pre-dwconv q/k/v, this head
    __shared__ float    dwWs[3][DHEAD][9];
    __shared__ float    dwBs[3][DHEAD];
    __shared__ _Float16 qs[NTOK][40];
    __shared__ _Float16 ks[NTOK][40];
    __shared__ _Float16 vt[DHEAD][72];         // V transposed [dim][token]
    __shared__ _Float16 ps[NTOK][72];          // softmax probs

    int lane = threadIdx.x;
    int h   = blockIdx.x & 7;
    int wid = blockIdx.x >> 3;
    int bimg = wid >> 8, rem = wid & 255;
    int wy = rem >> 4, wx = rem & 15;
    size_t rowbase = (size_t)bimg * LTOK + (size_t)(wy * WIN) * WWID + (size_t)(wx * WIN);
    int hd0 = h * DHEAD;

    // stage raw qkv channels of this head via async global->LDS
#pragma unroll
    for (int it = 0; it < 24; it++) {
        int idx = it * 32 + lane;            // 0..767 16B chunks
        int mat = idx >> 8; int r = idx & 255;
        int token = r >> 2; int ch = (r & 3) * 8;
        size_t grow = rowbase + (size_t)(token >> 3) * WWID + (token & 7);
        async_cp16(qkv + grow * 768 + mat * DIMC + hd0 + ch, &raw[mat][token][ch]);
    }
    // stage depthwise weights / bias for this head's channels
    for (int it = 0; it < 27; it++) {
        int idx = it * 32 + lane;            // 0..863
        int mat = idx / 288, r2 = idx % 288;
        int dd = r2 / 9, tap = r2 % 9;
        dwWs[mat][dd][tap] = dww[(size_t)(mat * DIMC + hd0 + dd) * 9 + tap];
    }
    for (int it = 0; it < 3; it++) {
        int idx = it * 32 + lane;            // 0..95
        int mat = idx >> 5, dd = idx & 31;
        dwBs[mat][dd] = dwb[mat * DIMC + hd0 + dd];
    }
    async_wait0();
    __syncthreads();

    float tmul = temp[h];
    float qsc  = 0.1767766953f * tmul;       // d^-0.5 * temperature

    // depthwise conv (zero pad at window edges) + L2 normalize q,k
#pragma unroll
    for (int tt = 0; tt < 2; tt++) {
        int token = lane + tt * 32;
        int iy = token >> 3, ix = token & 7;
        float ssq = 0.f, ssk = 0.f;
        for (int dd = 0; dd < DHEAD; dd++) {
            float aq = dwBs[0][dd], ak = dwBs[1][dd], av = dwBs[2][dd];
#pragma unroll
            for (int ky = 0; ky < 3; ky++)
#pragma unroll
                for (int kx = 0; kx < 3; kx++) {
                    int ny = iy + ky - 1, nx = ix + kx - 1;
                    if (ny < 0 || ny > 7 || nx < 0 || nx > 7) continue;
                    int nt = ny * 8 + nx;
                    aq += dwWs[0][dd][ky * 3 + kx] * (float)raw[0][nt][dd];
                    ak += dwWs[1][dd][ky * 3 + kx] * (float)raw[1][nt][dd];
                    av += dwWs[2][dd][ky * 3 + kx] * (float)raw[2][nt][dd];
                }
            ssq += aq * aq; ssk += ak * ak;
            qs[token][dd] = (_Float16)aq;
            ks[token][dd] = (_Float16)ak;
            vt[dd][token] = (_Float16)av;
        }
        float qn = rsqrtf(fmaxf(ssq, 1e-24f)) * qsc;
        float kn = rsqrtf(fmaxf(ssk, 1e-24f));
        for (int dd = 0; dd < DHEAD; dd++) {
            qs[token][dd] = (_Float16)((float)qs[token][dd] * qn);
            ks[token][dd] = (_Float16)((float)ks[token][dd] * kn);
        }
    }
    __syncthreads();

    int hl = lane >> 4, mr = lane & 15;

    // S = q * k^T : 4x4 tiles of 16x16, K = 32 (head dim) -> one WMMA each
    v16h qf[4], kf[4];
#pragma unroll
    for (int i = 0; i < 4; i++) {
        qf[i] = ldfrag(&qs[i * 16 + mr][0], hl);
        kf[i] = ldfrag(&ks[i * 16 + mr][0], hl);
    }
    v8f S[4][4] = {};
#pragma unroll
    for (int i = 0; i < 4; i++)
#pragma unroll
        for (int j = 0; j < 4; j++)
            S[i][j] = __builtin_amdgcn_wmma_f32_16x16x32_f16(
                false, qf[i], false, kf[j], (short)0, S[i][j], false, false);

    // + relative position bias, then row softmax (row = query token)
#pragma unroll
    for (int i = 0; i < 4; i++)
#pragma unroll
        for (int r = 0; r < 8; r++) {
            int qm = i * 16 + hl * 8 + r;
            int qy = qm >> 3, qx = qm & 7;
#pragma unroll
            for (int j = 0; j < 4; j++) {
                int kn = j * 16 + mr;
                int ky = kn >> 3, kx = kn & 7;
                int ridx = (qy - ky + 7) * 15 + (qx - kx + 7);
                S[i][j][r] += rpb[ridx * NHEADS + h];
            }
            float mx = -1e30f;
#pragma unroll
            for (int j = 0; j < 4; j++) mx = fmaxf(mx, S[i][j][r]);
#pragma unroll
            for (int o = 8; o > 0; o >>= 1) mx = fmaxf(mx, __shfl_xor(mx, o, 32));
            float sum = 0.f;
#pragma unroll
            for (int j = 0; j < 4; j++) { float e = __expf(S[i][j][r] - mx); S[i][j][r] = e; sum += e; }
#pragma unroll
            for (int o = 8; o > 0; o >>= 1) sum += __shfl_xor(sum, o, 32);
            float inv = 1.f / sum;
#pragma unroll
            for (int j = 0; j < 4; j++)
                ps[qm][j * 16 + mr] = (_Float16)(S[i][j][r] * inv);
        }
    __syncthreads();

    // O = P * V : 4x2 tiles of 16x16, K = 64 as two chained K=32 WMMAs
    v8f O[4][2] = {};
#pragma unroll
    for (int i = 0; i < 4; i++)
#pragma unroll
        for (int kk = 0; kk < 2; kk++) {
            v16h pf = ldfrag(&ps[i * 16 + mr][kk * 32], hl);
#pragma unroll
            for (int j = 0; j < 2; j++) {
                v16h vf = ldfrag(&vt[j * 16 + mr][kk * 32], hl);
                O[i][j] = __builtin_amdgcn_wmma_f32_16x16x32_f16(
                    false, pf, false, vf, (short)0, O[i][j], false, false);
            }
        }
    // store O: token-major, channel = h*32 + dim
#pragma unroll
    for (int i = 0; i < 4; i++)
#pragma unroll
        for (int j = 0; j < 2; j++)
#pragma unroll
            for (int r = 0; r < 8; r++) {
                int token = i * 16 + hl * 8 + r;
                int dd = j * 16 + mr;
                size_t grow = rowbase + (size_t)(token >> 3) * WWID + (token & 7);
                attno[grow * DIMC + hd0 + dd] = (_Float16)O[i][j][r];
            }
}

// ---------------------------------------------------------------------------
// LeFF depthwise 3x3 + GELU gate, spatially tiled:
// block = 16x16 token tile x 16 channels (both halves), halo 18x18 in LDS.
// Cuts the 9x global re-read to ~1.27x.
// ---------------------------------------------------------------------------
#define DGT 16   // tile side (tokens)
#define CCH 16   // channels per block (per half)
__global__ __launch_bounds__(256) void k_dwglu(const _Float16* __restrict__ pin,
                                               const float* __restrict__ w,
                                               const float* __restrict__ b,
                                               _Float16* __restrict__ g) {
    __shared__ _Float16 sh[2][DGT + 2][DGT + 2][CCH];  // 20.25 KB
    __shared__ float    sw[2][CCH][9];
    __shared__ float    sb[2][CCH];

    int blk = blockIdx.x;
    int cchunk = blk & 63;             // 64 channel chunks of 16
    int tile = blk >> 6;               // 0..255 = bimg*64 + ty*8 + tx
    int bimg = tile >> 6;
    int tr = tile & 63;
    int ty0 = (tr >> 3) * DGT, tx0 = (tr & 7) * DGT;
    int c0 = cchunk * CCH;
    int t = threadIdx.x;

    // stage dw weights / bias for these channels
    for (int i = t; i < 2 * CCH * 9; i += 256) {
        int hf = i / (CCH * 9), r = i % (CCH * 9);
        int cc = r / 9, tap = r % 9;
        sw[hf][cc][tap] = w[(size_t)(hf * HIDD + c0 + cc) * 9 + tap];
    }
    if (t < 2 * CCH) {
        int hf = t >> 4, cc = t & 15;
        sb[hf][cc] = b[hf * HIDD + c0 + cc];
    }
    // stage 18x18 halo, both halves: 18*18*2 halves*2 subchunks = 1296 chunks
    for (int i = t; i < 1296; i += 256) {
        int hf = (i >= 648) ? 1 : 0;
        int r = i - hf * 648;
        int tok = r >> 1;
        int sub = (r & 1) * 8;
        int yy = tok / 18, xx = tok % 18;
        int gy = ty0 + yy - 1, gx = tx0 + xx - 1;
        v8h val = {};
        if (gy >= 0 && gy < HH && gx >= 0 && gx < WWID)
            val = *(const v8h*)(pin + ((size_t)bimg * LTOK + (size_t)gy * WWID + gx) * (2 * HIDD)
                                + hf * HIDD + c0 + sub);
        *(v8h*)&sh[hf][yy][xx][sub] = val;
    }
    __syncthreads();

    int ty = t >> 4, tx = t & 15;
    float a1[CCH], a2[CCH];
#pragma unroll
    for (int e = 0; e < CCH; e++) { a1[e] = sb[0][e]; a2[e] = sb[1][e]; }
#pragma unroll
    for (int ky = 0; ky < 3; ky++)
#pragma unroll
        for (int kx = 0; kx < 3; kx++) {
            int tap = ky * 3 + kx;
            const _Float16* p1 = &sh[0][ty + ky][tx + kx][0];
            const _Float16* p2 = &sh[1][ty + ky][tx + kx][0];
            v8h u10 = *(const v8h*)(p1), u11 = *(const v8h*)(p1 + 8);
            v8h u20 = *(const v8h*)(p2), u21 = *(const v8h*)(p2 + 8);
#pragma unroll
            for (int e = 0; e < 8; e++) {
                a1[e]     += sw[0][e][tap]     * (float)u10[e];
                a1[e + 8] += sw[0][e + 8][tap] * (float)u11[e];
                a2[e]     += sw[1][e][tap]     * (float)u20[e];
                a2[e + 8] += sw[1][e + 8][tap] * (float)u21[e];
            }
        }
    v8h o0, o1;
#pragma unroll
    for (int e = 0; e < 8; e++) {
        float t0 = a1[e];
        float t1 = a1[e + 8];
        o0[e] = (_Float16)(0.5f * t0 * (1.f + erff(t0 * 0.7071067811865475f)) * a2[e]);
        o1[e] = (_Float16)(0.5f * t1 * (1.f + erff(t1 * 0.7071067811865475f)) * a2[e + 8]);
    }
    size_t grow = (size_t)bimg * LTOK + (size_t)(ty0 + ty) * WWID + (tx0 + tx);
    _Float16* dst = g + grow * HIDD + c0;
    *(v8h*)(dst)     = o0;
    *(v8h*)(dst + 8) = o1;
}

// ---------------------------------------------------------------------------
extern "C" void kernel_launch(void* const* d_in, const int* in_sizes, int n_in,
                              void* d_out, int out_size, void* d_ws, size_t ws_size,
                              hipStream_t stream) {
    (void)in_sizes; (void)n_in; (void)out_size; (void)ws_size;
    const float* x       = (const float*)d_in[0];
    const float* n1w     = (const float*)d_in[1];
    const float* n1b     = (const float*)d_in[2];
    const float* qkv_w   = (const float*)d_in[3];
    const float* qkv_b   = (const float*)d_in[4];
    const float* qkvdw_w = (const float*)d_in[5];
    const float* qkvdw_b = (const float*)d_in[6];
    const float* rpb     = (const float*)d_in[7];
    const float* temp    = (const float*)d_in[8];
    const float* proj_w  = (const float*)d_in[9];
    const float* proj_b  = (const float*)d_in[10];
    const float* n2w     = (const float*)d_in[11];
    const float* n2b     = (const float*)d_in[12];
    const float* pin_w   = (const float*)d_in[13];
    const float* pin_b   = (const float*)d_in[14];
    const float* dw_w    = (const float*)d_in[15];
    const float* dw_b    = (const float*)d_in[16];
    const float* pout_w  = (const float*)d_in[17];
    const float* pout_b  = (const float*)d_in[18];

    char* ws = (char*)d_ws;
    size_t off = 0;
    _Float16* wqkv  = (_Float16*)(ws + off); off += (size_t)768 * 256 * 2;
    _Float16* wproj = (_Float16*)(ws + off); off += (size_t)256 * 256 * 2;
    _Float16* wpin  = (_Float16*)(ws + off); off += (size_t)2048 * 256 * 2;
    _Float16* wpout = (_Float16*)(ws + off); off += (size_t)256 * 1024 * 2;
    _Float16* xn    = (_Float16*)(ws + off); off += (size_t)MROWS * 256 * 2;
    _Float16* qkvh  = (_Float16*)(ws + off); off += (size_t)MROWS * 768 * 2;
    _Float16* attno = (_Float16*)(ws + off); off += (size_t)MROWS * 256 * 2;
    _Float16* x1n   = (_Float16*)(ws + off); off += (size_t)MROWS * 256 * 2;
    _Float16* pinh  = (_Float16*)(ws + off); off += (size_t)MROWS * 2048 * 2;
    _Float16* gh    = (_Float16*)(ws + off); off += (size_t)MROWS * 1024 * 2;

    float* out = (float*)d_out;

    // weight conversions
    k_f32_to_f16<<<(768 * 256 + 255) / 256, 256, 0, stream>>>(qkv_w, wqkv, 768 * 256);
    k_f32_to_f16<<<(256 * 256 + 255) / 256, 256, 0, stream>>>(proj_w, wproj, 256 * 256);
    k_f32_to_f16<<<(2048 * 256 + 255) / 256, 256, 0, stream>>>(pin_w, wpin, 2048 * 256);
    k_f32_to_f16<<<(256 * 1024 + 255) / 256, 256, 0, stream>>>(pout_w, wpout, 256 * 1024);

    // norm1 -> f16
    k_ln<<<MROWS / 8, 256, 0, stream>>>(x, n1w, n1b, xn);

    // QKV 1x1 conv GEMM: 65536 x 768 x 256 -> f16
    {
        dim3 grid(768 / TBN, MROWS / TBM);
        k_gemm<true, false><<<grid, 256, 0, stream>>>(xn, wqkv, qkv_b, (void*)qkvh,
                                                      nullptr, MROWS, 768, 256);
    }
    // window attention (dwconv + cosine attn), one wave per (window, head)
    k_attn<<<BATCH * 256 * NHEADS, 32, 0, stream>>>(qkvh, qkvdw_w, qkvdw_b, rpb, temp, attno);

    // proj GEMM + residual(x) -> d_out holds x1 (f32)
    {
        dim3 grid(256 / TBN, MROWS / TBM);
        k_gemm<false, true><<<grid, 256, 0, stream>>>(attno, wproj, proj_b, (void*)out,
                                                      x, MROWS, 256, 256);
    }
    // norm2 -> f16
    k_ln<<<MROWS / 8, 256, 0, stream>>>(out, n2w, n2b, x1n);

    // LeFF pin GEMM: 65536 x 2048 x 256 -> f16
    {
        dim3 grid(2048 / TBN, MROWS / TBM);
        k_gemm<true, false><<<grid, 256, 0, stream>>>(x1n, wpin, pin_b, (void*)pinh,
                                                      nullptr, MROWS, 2048, 256);
    }
    // depthwise 3x3 + GELU gate (tiled): 256 tiles * 64 channel-chunks
    k_dwglu<<<BATCH * 64 * 64, 256, 0, stream>>>(pinh, dw_w, dw_b, gh);

    // LeFF pout GEMM + residual(x1 in d_out): 65536 x 256 x 1024 -> f32
    {
        dim3 grid(256 / TBN, MROWS / TBM);
        k_gemm<false, true><<<grid, 256, 0, stream>>>(gh, wpout, pout_b, (void*)out,
                                                      out, MROWS, 256, 1024);
    }
}
